// TwoDimDDSPGenerator_40450001994076
// MI455X (gfx1250) — matrix operand order repaired and checked by reading, not compiled, not run on hardware
//
#include <hip/hip_runtime.h>
#include <hip/hip_bf16.h>
#include <math.h>

// ---------------------------------------------------------------------------
// Types for CDNA5 WMMA (wave32, 16x16x32 f16 -> f32 accumulate) and TDM
// ---------------------------------------------------------------------------
typedef __attribute__((ext_vector_type(16))) _Float16 v16h;
typedef __attribute__((ext_vector_type(8)))  _Float16 v8h;
typedef __attribute__((ext_vector_type(8)))  float    v8f;
typedef __attribute__((ext_vector_type(4)))  unsigned int v4u;
typedef __attribute__((ext_vector_type(8)))  int          v8i;
typedef __attribute__((ext_vector_type(4)))  int          v4i;

#if defined(__has_builtin)
#if __has_builtin(__builtin_amdgcn_tensor_load_to_lds) && \
    __has_builtin(__builtin_amdgcn_s_wait_tensorcnt)
#define USE_TDM 1
#endif
#endif
#ifndef USE_TDM
#define USE_TDM 0
#endif

#define SRATE     11025.0f
#define TWO_PI_F  6.28318530717958647692f
#define UP_LEN    131072          // 512 * 256
#define N_ROWS    512             // B(4) * N_OSC(128)
#define N_CHUNK   512             // UP_LEN / 256

// ---------------------------------------------------------------------------
// Weight packing: f32 OIHW -> f16 [CoPad][Kpad], zero padded
// ---------------------------------------------------------------------------
__global__ void prep_weights(const float* __restrict__ w, _Float16* __restrict__ o,
                             int Co, int Kreal, int Kpad, int tot)
{
    int idx = blockIdx.x * 256 + threadIdx.x;
    if (idx >= tot) return;
    int co = idx / Kpad;
    int k  = idx - co * Kpad;
    float v = (co < Co && k < Kreal) ? w[(size_t)co * Kreal + k] : 0.0f;
    o[idx] = (_Float16)v;
}

__global__ void cast_f2h(const float* __restrict__ in, _Float16* __restrict__ o, int n)
{
    int i = blockIdx.x * 256 + threadIdx.x;
    if (i < n) o[i] = (_Float16)in[i];
}

// ---------------------------------------------------------------------------
// Implicit-GEMM conv via V_WMMA_F32_16X16X32_F16.
// Block = 8 waves = 128 contiguous outputs of one (b,ho) row x 16 out-channels.
// Weight panel [16][K-chunk] staged in LDS by the Tensor Data Mover
// (tensor_load_to_lds, TENSORcnt) when available, else cooperative copy.
// A fragment (weights, from LDS): lane<16 -> M=lane, K=kb+0..15
//                                 lane>=16 -> M=lane-16, K=kb+16..31
// B fragment (im2col, global):    lane<16 -> N=lane, K=kb+0..15
//                                 lane>=16 -> N=lane-16, K=kb+16..31
// C/D: VGPR r -> M = r + (lane>=16 ? 8:0), N = lane&15
// ---------------------------------------------------------------------------
template<int CI, int KH, int KW, int PH, int PW, int SH>
__global__ __launch_bounds__(256)
void conv_wmma(const _Float16* __restrict__ in, const _Float16* __restrict__ wk,
               const float* __restrict__ bias,
               _Float16* __restrict__ obf, float* __restrict__ of32,
               int Co, int Hin, int Win, int Hout, int Wout, int relu)
{
    constexpr int KREAL = CI * KH * KW;
    constexpr int KPAD  = (KREAL + 31) & ~31;
    constexpr int KC    = (KPAD < 512) ? KPAD : 512;   // K chunk staged in LDS

    __shared__ __align__(16) _Float16 wtile[16 * KC];  // 16 KB max

    const int tid  = threadIdx.x;
    const int lane = tid & 31;
    const int wave = tid >> 5;

    // strip decomposition: 128 contiguous W outputs of one (b, ho) row
    const int spr = Wout >> 7;                 // strips per row (W % 128 == 0)
    const int wo0 = (blockIdx.x % spr) << 7;
    const int hb  = blockIdx.x / spr;
    const int ho  = hb % Hout;                 // block-uniform (SALU)
    const int b   = hb / Hout;
    const int m0  = blockIdx.y * 16;
    const int hibase = ho * SH - PH;
    const _Float16* inB = in + (size_t)b * CI * Hin * Win;

    const int wo  = wo0 + wave * 16 + (lane & 15);   // this lane's output column
    const int bko = (lane & 16) ? 16 : 0;

    v8f acc = {0.f, 0.f, 0.f, 0.f, 0.f, 0.f, 0.f, 0.f};

    for (int kc0 = 0; kc0 < KPAD; kc0 += KC) {
        const int kclen = (KPAD - kc0 < KC) ? (KPAD - kc0) : KC;  // multiple of 32
        __syncthreads();
#if USE_TDM
        if (wave == 0) {
            // Tensor DMA: 2D tile, tile_dim0 = kclen elems, tile_dim1 = 16 rows,
            // row stride = KPAD elems; always in-bounds (weights zero-padded).
            unsigned long long ga =
                (unsigned long long)(const void*)(wk + (size_t)m0 * KPAD + kc0);
            unsigned int lds = (unsigned int)(unsigned long long)(void*)wtile;
            v4u g0;
            g0[0] = 1u;                                   // count=1, user descriptor
            g0[1] = lds;                                  // LDS byte address
            g0[2] = (unsigned int)ga;                     // global addr lo
            g0[3] = (unsigned int)((ga >> 32) & 0x1FFFFFFull) | 0x80000000u; // type=2
            unsigned int td0 = (unsigned int)kclen;       // tensor_dim0 (elems)
            unsigned int td1 = 16u;                       // tensor_dim1 (rows)
            unsigned long long st0 = (unsigned long long)KPAD; // dim0 stride (elems)
            v8i g1;
            g1[0] = 0x00010000;                           // data_size=1 (2 bytes)
            g1[1] = (int)((td0 & 0xFFFFu) << 16);         // tensor_dim0 lo16
            g1[2] = (int)(((td0 >> 16) & 0xFFFFu) | ((td1 & 0xFFFFu) << 16));
            g1[3] = (int)(((td1 >> 16) & 0xFFFFu) | ((unsigned)kclen << 16)); // tile_dim0
            g1[4] = (int)16;                              // tile_dim1=16, tile_dim2=0
            g1[5] = (int)(unsigned int)(st0 & 0xFFFFFFFFull);
            g1[6] = (int)(unsigned int)((st0 >> 32) & 0xFFFFull);
            g1[7] = 0;
            v4i gz  = {0, 0, 0, 0};                       // groups 2/3 unused (2D)
            v8i gz8 = {0, 0, 0, 0, 0, 0, 0, 0};           // extra descriptor words
            __builtin_amdgcn_tensor_load_to_lds(g0, g1, gz, gz, gz8, 0);
            __builtin_amdgcn_s_wait_tensorcnt(0);
        }
#else
        for (int idx = tid; idx < 16 * kclen; idx += 256) {
            int r = idx / kclen;
            int c = idx - r * kclen;
            wtile[idx] = wk[(size_t)(m0 + r) * KPAD + kc0 + c];
        }
#endif
        __syncthreads();

        const _Float16* wr = wtile + (lane & 15) * kclen + ((lane & 16) ? 16 : 0);
        for (int kb = 0; kb < kclen; kb += 32) {
            // A fragment: two 16B LDS vector loads (ds_load_b128)
            v8h alo = *(const v8h*)(wr + kb);
            v8h ahi = *(const v8h*)(wr + kb + 8);
            v16h a;
#pragma unroll
            for (int i = 0; i < 8; i++) { a[i] = alo[i]; a[8 + i] = ahi[i]; }

            // B fragment: im2col gather, incremental (ci,kh,kw) decomposition
            const int k0 = kc0 + kb + bko;
            int ci = k0 / (KH * KW);
            int rr = k0 - ci * (KH * KW);
            int kh = rr / KW;
            int kw = rr - kh * KW;
            v16h bm;
#pragma unroll
            for (int i = 0; i < 16; i++) {
                _Float16 v = (_Float16)0.f;
                if (k0 + i < KREAL) {
                    const int hi = hibase + kh;           // uniform across lanes
                    const int wi = wo - PW + kw;
                    if (hi >= 0 && hi < Hin && wi >= 0 && wi < Win)
                        v = inB[((size_t)ci * Hin + hi) * Win + wi];
                }
                bm[i] = v;
                kw++;
                if (kw == KW) { kw = 0; kh++; if (kh == KH) { kh = 0; ci++; } }
            }
            acc = __builtin_amdgcn_wmma_f32_16x16x32_f16(
                      false, a, false, bm, (short)0, acc, false, false);
        }
    }

    // Epilogue: bias + leaky-relu, store f16 (next layer) and/or f32 (heads)
    const int mb = m0 + ((lane & 16) ? 8 : 0);
#pragma unroll
    for (int r = 0; r < 8; r++) {
        const int m = mb + r;
        if (m < Co) {
            float v = acc[r] + bias[m];
            if (relu && v < 0.f) v *= 0.2f;
            const size_t oi = (((size_t)b * Co + m) * Hout + ho) * Wout + wo;
            if (obf)  obf[oi]  = (_Float16)v;
            if (of32) of32[oi] = v;
        }
    }
}

// ---------------------------------------------------------------------------
// tp head: l = p0^2 ; f = centers + tanh(p1)*0.5*erbs    (B,128,512)
// ---------------------------------------------------------------------------
__global__ void head_kernel(const float* __restrict__ tp,
                            const float* __restrict__ centers,
                            const float* __restrict__ erbs,
                            float* __restrict__ l512, float* __restrict__ f512)
{
    int idx = blockIdx.x * 256 + threadIdx.x;
    if (idx >= 4 * 128 * 512) return;
    int t = idx % 512;
    int o = (idx / 512) & 127;
    int b = idx / (512 * 128);
    float p0 = tp[(((size_t)b * 2 + 0) * 128 + o) * 512 + t];
    float p1 = tp[(((size_t)b * 2 + 1) * 128 + o) * 512 + t];
    l512[idx] = p0 * p0;
    f512[idx] = centers[o] + tanhf(p1) * 0.5f * erbs[o];
}

// ---------------------------------------------------------------------------
// linear resize along W, f16->f16 (noise branch: 512 -> 8192)
// ---------------------------------------------------------------------------
__global__ void resize_w16(const _Float16* __restrict__ in, _Float16* __restrict__ out,
                           int rows, int Wi, int Wo)
{
    int idx = blockIdx.x * 256 + threadIdx.x;
    if (idx >= rows * Wo) return;
    int t = idx % Wo;
    int r = idx / Wo;
    float src = (t + 0.5f) * ((float)Wi / (float)Wo) - 0.5f;
    float fj  = floorf(src);
    int   j   = (int)fj;
    float w   = src - fj;
    int j0 = j < 0 ? 0 : (j > Wi - 1 ? Wi - 1 : j);
    int j1 = j + 1 < 0 ? 0 : (j + 1 > Wi - 1 ? Wi - 1 : j + 1);
    float v = (float)in[(size_t)r * Wi + j0] * (1.f - w) + (float)in[(size_t)r * Wi + j1] * w;
    out[idx] = (_Float16)v;
}

// ---------------------------------------------------------------------------
// Oscillator bank helpers: half-pixel linear interp 512 -> 131072 (factor 256)
// ---------------------------------------------------------------------------
__device__ inline float interp512(const float* __restrict__ rowp, int t)
{
    float src = (t + 0.5f) * (1.0f / 256.0f) - 0.5f;
    float fj  = floorf(src);
    int   j   = (int)fj;
    float w   = src - fj;
    int j0 = j < 0 ? 0 : (j > 511 ? 511 : j);
    int j1 = j + 1 < 0 ? 0 : (j + 1 > 511 ? 511 : j + 1);
    return rowp[j0] * (1.f - w) + rowp[j1] * w;
}

__device__ inline float osc_df(const float* __restrict__ frow, int t)
{
    float f = interp512(frow, t);
    f = fminf(fmaxf(f, 20.0f), 5512.5f);
    return f * (TWO_PI_F / SRATE);
}

// Phase scan stage 1: per-(row, 256-sample chunk) sums
__global__ __launch_bounds__(256)
void osc_chunk_sums(const float* __restrict__ f512, float* __restrict__ cs)
{
    int chunk = blockIdx.x;           // 0..511
    int row   = blockIdx.y;           // 0..511
    int t     = chunk * 256 + threadIdx.x;
    float v = osc_df(f512 + (size_t)row * 512, t);
    for (int d = 16; d; d >>= 1) v += __shfl_down(v, d, 32);
    __shared__ float wsum[8];
    int lane = threadIdx.x & 31, wv = threadIdx.x >> 5;
    if (lane == 0) wsum[wv] = v;
    __syncthreads();
    if (threadIdx.x == 0) {
        float s = 0.f;
        for (int i = 0; i < 8; i++) s += wsum[i];
        cs[(size_t)row * N_CHUNK + chunk] = s;
    }
}

// Phase scan stage 2: per-row exclusive scan of 512 chunk sums (1 wave/row)
__global__ __launch_bounds__(32)
void osc_scan(const float* __restrict__ cs, float* __restrict__ cp)
{
    int row  = blockIdx.x;
    int lane = threadIdx.x;
    float carry = 0.f;
    for (int i = 0; i < 16; i++) {
        float v = cs[(size_t)row * N_CHUNK + i * 32 + lane];
        float s = v;
        for (int d = 1; d < 32; d <<= 1) { float o = __shfl_up(s, d, 32); if (lane >= d) s += o; }
        cp[(size_t)row * N_CHUNK + i * 32 + lane] = carry + s - v;   // exclusive
        carry += __shfl(s, 31, 32);
    }
}

// Phase scan stage 3 + oscillator sum. One block = (chunk, batch); loops all
// 128 oscillators accumulating in registers -> deterministic, no atomics.
__global__ __launch_bounds__(256)
void osc_accum(const float* __restrict__ f512, const float* __restrict__ l512,
               const float* __restrict__ cp, float* __restrict__ harm)
{
    int chunk = blockIdx.x;           // 0..511
    int b     = blockIdx.y;           // 0..3
    int t     = chunk * 256 + threadIdx.x;
    int lane  = threadIdx.x & 31, wv = threadIdx.x >> 5;
    __shared__ float wt[8];
    float acc = 0.f;
    for (int o = 0; o < 128; o++) {
        int row = b * 128 + o;
        float v = osc_df(f512 + (size_t)row * 512, t);
        float s = v;
        for (int d = 1; d < 32; d <<= 1) { float x = __shfl_up(s, d, 32); if (lane >= d) s += x; }
        if (lane == 31) wt[wv] = s;
        __syncthreads();
        float off = 0.f;
#pragma unroll
        for (int i = 0; i < 8; i++) if (i < wv) off += wt[i];
        float phase = cp[(size_t)row * N_CHUNK + chunk] + off + s;   // inclusive cumsum
        float amp   = interp512(l512 + (size_t)row * 512, t);
        acc += __sinf(phase) * amp;
        __syncthreads();
    }
    harm[(size_t)b * UP_LEN + t] = acc;
}

// ---------------------------------------------------------------------------
// noise_bank2: per (b, segment s) one wave.
// filt = irfft(rfft(frame)*mags) == circular conv with
//   g[d] = (1/32) * sum_k w_k * mag_k * cos(2*pi*k*d/32), w_0=w_16=1 else 2
// out[s*16+j] = filt_s[j] + filt_{s-1}[16+j]
// ---------------------------------------------------------------------------
__global__ __launch_bounds__(32)
void noise_ola(const float* __restrict__ noise, const float* __restrict__ n4,
               float* __restrict__ nout)
{
    int s = blockIdx.x & 8191;
    int b = blockIdx.x >> 13;
    int lane = threadIdx.x;
    __shared__ float xs[32], xs1[32], gs[32], gs1[32], tmp[32];

    const float* nb = noise + (size_t)b * UP_LEN;
    int i0 = s * 16 + lane;
    xs[lane]  = (i0 < UP_LEN) ? nb[i0] : 0.0f;                 // hop zero-pad
    xs1[lane] = (s > 0) ? nb[(s - 1) * 16 + lane] : 0.0f;      // frame s-1 (OLA)

    const float* m4 = n4 + (size_t)b * 17 * 8192;
    float g0 = 0.f, g1 = 0.f;
    for (int k = 0; k < 17; k++) {
        float wk = (k == 0 || k == 16) ? 1.0f : 2.0f;
        float c  = __cosf((TWO_PI_F / 32.0f) * (float)((k * lane) & 31));
        float mk = m4[(size_t)k * 8192 + s];  mk *= mk;        // mags = n^2
        g0 += wk * mk * c;
        if (s > 0) { float mp = m4[(size_t)k * 8192 + s - 1]; mp *= mp; g1 += wk * mp * c; }
    }
    gs[lane]  = g0 * (1.0f / 32.0f);
    gs1[lane] = g1 * (1.0f / 32.0f);
    __syncthreads();

    float acc = 0.f;
    if (lane < 16) { for (int m = 0; m < 32; m++) acc += xs[m]  * gs [(lane - m) & 31]; }
    else           { for (int m = 0; m < 32; m++) acc += xs1[m] * gs1[(lane - m) & 31]; }
    tmp[lane] = acc;
    __syncthreads();
    if (lane < 16)
        nout[(size_t)b * UP_LEN + s * 16 + lane] = tmp[lane] + tmp[lane + 16];
}

// final: out = (harmonic + noise)[:, :, 4096:-4096]
__global__ void combine(const float* __restrict__ harm, const float* __restrict__ nout,
                        float* __restrict__ out)
{
    int idx = blockIdx.x * 256 + threadIdx.x;
    if (idx >= 4 * 122880) return;
    int t = idx % 122880;
    int b = idx / 122880;
    size_t src = (size_t)b * UP_LEN + 4096 + t;
    out[idx] = harm[src] + nout[src];
}

// ---------------------------------------------------------------------------
// Host launcher
// ---------------------------------------------------------------------------
extern "C" void kernel_launch(void* const* d_in, const int* in_sizes, int n_in,
                              void* d_out, int out_size, void* d_ws, size_t ws_size,
                              hipStream_t stream)
{
    (void)in_sizes; (void)n_in; (void)out_size; (void)ws_size;

    const float* x       = (const float*)d_in[0];
    const float* noise   = (const float*)d_in[1];
    const float* m1_w = (const float*)d_in[2];  const float* m1_b = (const float*)d_in[3];
    const float* m2_w = (const float*)d_in[4];  const float* m2_b = (const float*)d_in[5];
    const float* m3_w = (const float*)d_in[6];  const float* m3_b = (const float*)d_in[7];
    const float* m4_w = (const float*)d_in[8];  const float* m4_b = (const float*)d_in[9];
    const float* m5_w = (const float*)d_in[10]; const float* m5_b = (const float*)d_in[11];
    const float* tp_w = (const float*)d_in[12]; const float* tp_b = (const float*)d_in[13];
    const float* n1_w = (const float*)d_in[14]; const float* n1_b = (const float*)d_in[15];
    const float* n2_w = (const float*)d_in[16]; const float* n2_b = (const float*)d_in[17];
    const float* n3_w = (const float*)d_in[18]; const float* n3_b = (const float*)d_in[19];
    const float* n4_w = (const float*)d_in[20]; const float* n4_b = (const float*)d_in[21];
    const float* centers = (const float*)d_in[22];
    const float* erbs    = (const float*)d_in[23];
    float* out = (float*)d_out;

    // ---- workspace carve-up (~150 MB; activations ping-pong in f16) ----
    char* base = (char*)d_ws;
    size_t off = 0;
    auto take = [&](size_t bytes) -> char* {
        char* p = base + off;
        off += (bytes + 255) & ~(size_t)255;
        return p;
    };
    _Float16* R0   = (_Float16*)take((size_t)33554432 * 2);     // 4*64*256*512 f16
    _Float16* R1   = (_Float16*)take((size_t)33554432 * 2);
    float*    TP   = (float*)take((size_t)4 * 2 * 128 * 512 * 4);
    float*    N4   = (float*)take((size_t)4 * 17 * 8192 * 4);
    float*    L512 = (float*)take((size_t)4 * 128 * 512 * 4);
    float*    F512 = (float*)take((size_t)4 * 128 * 512 * 4);
    float*    CS   = (float*)take((size_t)N_ROWS * N_CHUNK * 4);
    float*    CP   = (float*)take((size_t)N_ROWS * N_CHUNK * 4);
    float*    HARM = (float*)take((size_t)4 * UP_LEN * 4);
    float*    NOUT = (float*)take((size_t)4 * UP_LEN * 4);
    _Float16* WGT  = (_Float16*)take((size_t)1 << 20);

    // packed weight sub-buffers [CoPad][Kpad], all offsets 256B aligned
    _Float16* Wm1 = WGT;            // 32 x 32
    _Float16* Wm2 = Wm1 + 1024;     // 32 x 864
    _Float16* Wm3 = Wm2 + 27648;    // 64 x 864
    _Float16* Wm4 = Wm3 + 55296;    // 64 x 1728
    _Float16* Wm5 = Wm4 + 110592;   // 32 x 1728
    _Float16* Wtp = Wm5 + 55296;    // 16 x 288
    _Float16* Wn1 = Wtp + 4608;     // 16 x 1632
    _Float16* Wn2 = Wn1 + 26112;    // 16 x 832
    _Float16* Wn3 = Wn2 + 13312;    // 16 x 96
    _Float16* Wn4 = Wn3 + 1536;     // 32 x 160

    auto prepw = [&](const float* w, _Float16* dst, int Co, int Ci, int KH, int KW, int CoPad) {
        int Kreal = Ci * KH * KW;
        int Kpad  = (Kreal + 31) & ~31;
        int tot   = CoPad * Kpad;
        prep_weights<<<(tot + 255) / 256, 256, 0, stream>>>(w, dst, Co, Kreal, Kpad, tot);
    };
    prepw(m1_w, Wm1, 32,  1,  3, 9, 32);
    prepw(m2_w, Wm2, 32, 32,  9, 3, 32);
    prepw(m3_w, Wm3, 64, 32,  3, 9, 64);
    prepw(m4_w, Wm4, 64, 64,  3, 9, 64);
    prepw(m5_w, Wm5, 32, 64,  3, 9, 32);
    prepw(tp_w, Wtp,  2, 32,  3, 3, 16);
    prepw(n1_w, Wn1, 16, 32, 17, 3, 16);
    prepw(n2_w, Wn2, 16, 16, 17, 3, 16);
    prepw(n3_w, Wn3, 16, 16,  2, 3, 16);
    prepw(n4_w, Wn4, 17, 16,  3, 3, 32);

    // input x -> f16
    cast_f2h<<<(524288 + 255) / 256, 256, 0, stream>>>(x, R0, 524288);

#define CONV(CI, KH, KW, PH, PW, SH, inp, wgt, bias, obf, of32, Co, CoPad, Hin, Win, Hout, Wout, relu) \
    do {                                                                                              \
        dim3 g_((4 * (Hout) * (Wout)) / 128, (CoPad) / 16);                                           \
        conv_wmma<CI, KH, KW, PH, PW, SH><<<g_, 256, 0, stream>>>(                                    \
            inp, wgt, bias, obf, of32, Co, Hin, Win, Hout, Wout, relu);                               \
    } while (0)

    // main conv stack (f16 ping-pong between R0 and R1)
    CONV( 1,  3, 9, 1, 4, 1, R0, Wm1, m1_b, R1, (float*)nullptr, 32, 32, 256, 512, 256, 512, 1);
    CONV(32,  9, 3, 4, 1, 1, R1, Wm2, m2_b, R0, (float*)nullptr, 32, 32, 256, 512, 256, 512, 1);
    CONV(32,  3, 9, 1, 4, 1, R0, Wm3, m3_b, R1, (float*)nullptr, 64, 64, 256, 512, 256, 512, 1);
    CONV(64,  3, 9, 1, 4, 2, R1, Wm4, m4_b, R0, (float*)nullptr, 64, 64, 256, 512, 128, 512, 1);
    CONV(64,  3, 9, 1, 4, 1, R0, Wm5, m5_b, R1, (float*)nullptr, 32, 32, 128, 512, 128, 512, 1);
    // tp head (f32 out, no relu); pre == R1
    CONV(32,  3, 3, 1, 1, 1, R1, Wtp, tp_b, (_Float16*)nullptr, TP, 2, 16, 128, 512, 128, 512, 0);
    // noise branch
    CONV(32, 17, 3, 8, 1, 8, R1, Wn1, n1_b, R0, (float*)nullptr, 16, 16, 128, 512, 16, 512, 1);
    resize_w16<<<(1024 * 8192 + 255) / 256, 256, 0, stream>>>(R0, R1, 1024, 512, 8192);
    CONV(16, 17, 3, 8, 1, 8, R1, Wn2, n2_b, R0, (float*)nullptr, 16, 16, 16, 8192, 2, 8192, 1);
    CONV(16,  2, 3, 0, 1, 1, R0, Wn3, n3_b, R1, (float*)nullptr, 16, 16, 2, 8192, 1, 8192, 1);
    CONV(16,  3, 3, 1, 1, 1, R1, Wn4, n4_b, (_Float16*)nullptr, N4, 17, 32, 1, 8192, 1, 8192, 0);

    // oscillator head
    head_kernel<<<(4 * 128 * 512 + 255) / 256, 256, 0, stream>>>(TP, centers, erbs, L512, F512);
    {
        dim3 g(N_CHUNK, N_ROWS);
        osc_chunk_sums<<<g, 256, 0, stream>>>(F512, CS);
    }
    osc_scan<<<N_ROWS, 32, 0, stream>>>(CS, CP);
    {
        dim3 g(N_CHUNK, 4);
        osc_accum<<<g, 256, 0, stream>>>(F512, L512, CP, HARM);
    }

    // noise bank + final mix
    noise_ola<<<4 * 8192, 32, 0, stream>>>(noise, N4, NOUT);
    combine<<<(4 * 122880 + 255) / 256, 256, 0, stream>>>(HARM, NOUT, out);

#undef CONV
}
